// ScaledDotProductAttention_89343909691909
// MI455X (gfx1250) — compile-verified, matrix operand order checked by reading, not run on hardware
//
#include <hip/hip_runtime.h>
#include <cstdint>

typedef _Float16 h16v __attribute__((ext_vector_type(16)));
typedef _Float16 h8v  __attribute__((ext_vector_type(8)));
typedef _Float16 h4v  __attribute__((ext_vector_type(4)));
typedef float    f8v  __attribute__((ext_vector_type(8)));
typedef float    f4v  __attribute__((ext_vector_type(4)));
typedef unsigned int v4u __attribute__((ext_vector_type(4)));
typedef int          v8i __attribute__((ext_vector_type(8)));
typedef int          v4i __attribute__((ext_vector_type(4)));

#define S_LEN   4096
#define DHEAD   64
#define NHEAD   16
#define BR      64              // q rows per block (16 per wave x 4 waves)
#define BC      64              // keys per tile
#define NKT     (S_LEN / BC)
#define LPAD    8
#define LDW     (DHEAD + LPAD)  // 72 halves = 144 B = 32 DW row + 4 DW pad (matches TDM pad cfg)

#define CAT16(lo, hi) __builtin_shufflevector((lo), (hi), 0,1,2,3,4,5,6,7,8,9,10,11,12,13,14,15)

// ---------------------------------------------------------------------------
// TDM: 2-D tile load Global->LDS. D# packed per ISA ch.8 (group0 128b, group1
// 256b; groups 2/3 + extra group zero for 2-D). data_size=2B; pad 4 DWORDs
// after every 32 DWORDs so a 64-half tile row lands on a 144-B padded LDS row.
// Toolchain here is clang-23 (therock): 6-arg builtin form.
// ---------------------------------------------------------------------------
__device__ __forceinline__ void tdm_load_2d(unsigned lds_addr, const _Float16* gptr,
                                            unsigned tensor_dim0, unsigned tensor_dim1,
                                            unsigned tile_dim0,   unsigned tile_dim1,
                                            unsigned dim0_stride /* elements */)
{
    unsigned long long ga = (unsigned long long)(size_t)gptr;
    v4u g0;
    g0[0] = 1u;                                               // count=1, user descriptor
    g0[1] = lds_addr;                                         // LDS byte address
    g0[2] = (unsigned)ga;                                     // global_addr[31:0]
    g0[3] = ((unsigned)(ga >> 32) & 0x01FFFFFFu) | (2u << 30);// global_addr[56:32] | type=2
    v8i g1;
    g1[0] = (int)((1u << 16) | (1u << 20) | (4u << 22) | (3u << 25)); // 2B, pad_en, int=32DW, amt=4DW
    g1[1] = (int)((tensor_dim0 & 0xFFFFu) << 16);                     // tensor_dim0[15:0]
    g1[2] = (int)((tensor_dim0 >> 16) | ((tensor_dim1 & 0xFFFFu) << 16));
    g1[3] = (int)((tensor_dim1 >> 16) | (tile_dim0 << 16));
    g1[4] = (int)tile_dim1;                                           // tile_dim2 = 0 (2-D)
    g1[5] = (int)dim0_stride;
    g1[6] = 0;                                                        // stride hi / dim1_stride
    g1[7] = 0;
    v4i z4 = (v4i)0;
    v8i z8 = (v8i)0;
    __builtin_amdgcn_tensor_load_to_lds(g0, g1, z4, z4, z8, 0);
}

// ---------------------------------------------------------------------------
// Pre-pass 1: K fp32 -> f16 (same [h][s][d] layout). One-time; removes the
// per-q-block re-conversion (64x redundant) and halves K's L2 traffic.
// ---------------------------------------------------------------------------
__global__ __launch_bounds__(256)
void cvt_f16_kernel(const float* __restrict__ src, _Float16* __restrict__ dst, int n8)
{
    int g = blockIdx.x * 256 + threadIdx.x;
    if (g >= n8) return;
    const float* p = src + (size_t)g * 8;
    f4v a = *(const f4v*)p;
    f4v b = *(const f4v*)(p + 4);
    h8v o;
#pragma unroll
    for (int e = 0; e < 4; ++e) { o[e] = (_Float16)a[e]; o[4 + e] = (_Float16)b[e]; }
    *(h8v*)(dst + (size_t)g * 8) = o;
}

// ---------------------------------------------------------------------------
// Pre-pass 2: V fp32 [h][s][d] -> f16 transposed [h][d][s] via LDS tile, so the
// main loop's V staging is a plain 2-D TDM copy into VsT[d][key].
// ---------------------------------------------------------------------------
__global__ __launch_bounds__(256)
void vtrans_kernel(const float* __restrict__ V, _Float16* __restrict__ VhT)
{
    __shared__ _Float16 T[64][72];
    const int tid = threadIdx.x;
    const int s0  = blockIdx.x * 64;
    const int h   = blockIdx.y;
#pragma unroll
    for (int i = 0; i < 4; ++i) {
        const int idx = i * 256 + tid;          // 1024 float4's per 64x64 tile
        const int r   = idx >> 4;
        const int c4  = (idx & 15) * 4;
        f4v v = *(const f4v*)(V + ((size_t)h * S_LEN + s0 + r) * DHEAD + c4);
#pragma unroll
        for (int e = 0; e < 4; ++e) T[c4 + e][r] = (_Float16)v[e];
    }
    __syncthreads();
#pragma unroll
    for (int i = 0; i < 2; ++i) {
        const int idx = i * 256 + tid;          // 512 h8v chunks out
        const int d   = idx >> 3;
        const int c8  = (idx & 7) * 8;
        *(h8v*)(VhT + ((size_t)h * DHEAD + d) * S_LEN + s0 + c8) = *(const h8v*)&T[d][c8];
    }
}

// ---------------------------------------------------------------------------
// Main flash-attention kernel. 4 waves x 16 q-rows; per 64-key tile:
// 8 WMMA (QK^T) + 2 WMMA (rowsum via ones) + 8 WMMA (PV). K/V tiles arrive by
// TDM into double-buffered LDS (s_wait_tensorcnt), overlapping DMA w/ compute.
// ---------------------------------------------------------------------------
__global__ __launch_bounds__(128)
void fa_wmma_kernel(const float* __restrict__ Q, const _Float16* __restrict__ Kh,
                    const _Float16* __restrict__ VhT, float* __restrict__ O)
{
    __shared__ _Float16 Ks [2][BC][LDW];     // [buf][key][d]
    __shared__ _Float16 VsT[2][DHEAD][LDW];  // [buf][d][key]
    __shared__ _Float16 Ps [BR][LDW];        // [q_local][key] (per-wave private strips)

    const int tid  = threadIdx.x;
    const int lane = tid & 31;
    const int wave = __builtin_amdgcn_readfirstlane(tid >> 5);  // SGPR (uniform TDM operands)
    const int half = lane >> 4;
    const int l16  = lane & 15;
    const int head = blockIdx.y;
    const int qw   = blockIdx.x * BR + wave * 16;

    const _Float16* KhH  = Kh  + (size_t)head * S_LEN * DHEAD;  // [s][d] f16
    const _Float16* VhTH = VhT + (size_t)head * DHEAD * S_LEN;  // [d][s] f16

    // ---- Q A-fragments (1/sqrt(64) folded in). A-map: lane=M&15, elem i ->
    // K = (i>>3)*16 + half*8 + (i&7) ----
    h16v qa[2];
    {
        const float  sc   = 0.125f;
        const float* qrow = Q + ((size_t)head * S_LEN + qw + l16) * DHEAD;
#pragma unroll
        for (int kc = 0; kc < 2; ++kc)
#pragma unroll
            for (int g = 0; g < 2; ++g) {
                const int dbase = kc * 32 + g * 16 + half * 8;
                f4v v0 = *(const f4v*)(qrow + dbase);
                f4v v1 = *(const f4v*)(qrow + dbase + 4);
#pragma unroll
                for (int e = 0; e < 4; ++e) {
                    qa[kc][g * 8 + e]     = (_Float16)(v0[e] * sc);
                    qa[kc][g * 8 + 4 + e] = (_Float16)(v1[e] * sc);
                }
            }
    }

    h16v ones;
#pragma unroll
    for (int i = 0; i < 16; ++i) ones[i] = (_Float16)1.0f;

    float m_run[8], l_run[8];
    f8v   acc[4];
#pragma unroll
    for (int j = 0; j < 8; ++j) { m_run[j] = -1e30f; l_run[j] = 0.0f; }
#pragma unroll
    for (int t = 0; t < 4; ++t)
#pragma unroll
        for (int j = 0; j < 8; ++j) acc[t][j] = 0.0f;

    // ---- Pre-issue TDM for tile 0: wave w DMAs K rows / V d-rows w*16..w*16+15
    tdm_load_2d((unsigned)(size_t)&Ks[0][wave * 16][0],
                KhH + (size_t)(wave * 16) * DHEAD,
                DHEAD, S_LEN, DHEAD, 16, DHEAD);
    tdm_load_2d((unsigned)(size_t)&VsT[0][wave * 16][0],
                VhTH + (size_t)(wave * 16) * S_LEN,
                S_LEN, DHEAD, BC, 16, S_LEN);

    for (int kt = 0; kt < NKT; ++kt) {
        const int buf = kt & 1;
        __builtin_amdgcn_s_wait_tensorcnt(0);   // this tile's DMA done (my waves' ops)
        __syncthreads();                        // all waves' quarters visible

        if (kt + 1 < NKT) {                     // kick next tile into the other buffer
            const int nb = buf ^ 1;
            tdm_load_2d((unsigned)(size_t)&Ks[nb][wave * 16][0],
                        KhH + (size_t)((kt + 1) * BC + wave * 16) * DHEAD,
                        DHEAD, S_LEN, DHEAD, 16, DHEAD);
            tdm_load_2d((unsigned)(size_t)&VsT[nb][wave * 16][0],
                        VhTH + (size_t)(wave * 16) * S_LEN + (kt + 1) * BC,
                        S_LEN, DHEAD, BC, 16, S_LEN);
        }

        // ---- S = (Q/sqrt d) K^T ----
        f8v s[4];
#pragma unroll
        for (int jb = 0; jb < 4; ++jb) {
            f8v c;
#pragma unroll
            for (int j = 0; j < 8; ++j) c[j] = 0.0f;
#pragma unroll
            for (int kc = 0; kc < 2; ++kc) {
                const _Float16* kr = &Ks[buf][jb * 16 + l16][0];   // B[k][n]=K[n][d]
                h8v lo = *(const h8v*)(kr + kc * 32 + half * 8);
                h8v hi = *(const h8v*)(kr + kc * 32 + 16 + half * 8);
                h16v kb = CAT16(lo, hi);
                c = __builtin_amdgcn_wmma_f32_16x16x32_f16(
                        false, qa[kc], false, kb, (short)0, c, false, false);
            }
            s[jb] = c;
        }

        // ---- Row max (butterfly within 16-lane halves), exp, rescale ----
        float mnew[8], scl[8];
#pragma unroll
        for (int j = 0; j < 8; ++j) {
            float t = fmaxf(fmaxf(s[0][j], s[1][j]), fmaxf(s[2][j], s[3][j]));
#pragma unroll
            for (int m = 1; m <= 8; m <<= 1) t = fmaxf(t, __shfl_xor(t, m, 32));
            mnew[j]  = fmaxf(m_run[j], t);
            scl[j]   = __expf(m_run[j] - mnew[j]);
            m_run[j] = mnew[j];
        }
#pragma unroll
        for (int jb = 0; jb < 4; ++jb)
#pragma unroll
            for (int j = 0; j < 8; ++j) s[jb][j] = __expf(s[jb][j] - mnew[j]);
#pragma unroll
        for (int db = 0; db < 4; ++db)
#pragma unroll
            for (int j = 0; j < 8; ++j) acc[db][j] *= scl[j];

        // ---- P -> per-wave LDS strip (f16), reload as A-fragments ----
        const int q0l = wave * 16;
#pragma unroll
        for (int jb = 0; jb < 4; ++jb)
#pragma unroll
            for (int j = 0; j < 8; ++j)
                Ps[q0l + j + half * 8][jb * 16 + l16] = (_Float16)s[jb][j];

        h16v pa[2];
#pragma unroll
        for (int kc = 0; kc < 2; ++kc) {
            const _Float16* pr = &Ps[q0l + l16][0];
            h8v lo = *(const h8v*)(pr + kc * 32 + half * 8);
            h8v hi = *(const h8v*)(pr + kc * 32 + 16 + half * 8);
            pa[kc] = CAT16(lo, hi);
        }

        // ---- Row sums via WMMA with ones (replaces 32 ds_bpermute) ----
        f8v rs;
#pragma unroll
        for (int j = 0; j < 8; ++j) rs[j] = 0.0f;
        rs = __builtin_amdgcn_wmma_f32_16x16x32_f16(false, pa[0], false, ones, (short)0, rs, false, false);
        rs = __builtin_amdgcn_wmma_f32_16x16x32_f16(false, pa[1], false, ones, (short)0, rs, false, false);
#pragma unroll
        for (int j = 0; j < 8; ++j) l_run[j] = l_run[j] * scl[j] + rs[j];

        // ---- O += P * V ----
#pragma unroll
        for (int kc = 0; kc < 2; ++kc)
#pragma unroll
            for (int db = 0; db < 4; ++db) {
                const _Float16* vr = &VsT[buf][db * 16 + l16][0];  // B[k][n]=V[k][d=n]
                h8v vlo = *(const h8v*)(vr + kc * 32 + half * 8);
                h8v vhi = *(const h8v*)(vr + kc * 32 + 16 + half * 8);
                h16v vb = CAT16(vlo, vhi);
                acc[db] = __builtin_amdgcn_wmma_f32_16x16x32_f16(
                              false, pa[kc], false, vb, (short)0, acc[db], false, false);
            }
    }

    // ---- Normalize + store O (fp32) ----
    const size_t obase = ((size_t)head * S_LEN + qw) * DHEAD;
#pragma unroll
    for (int db = 0; db < 4; ++db)
#pragma unroll
        for (int j = 0; j < 8; ++j) {
            const float o = acc[db][j] * (1.0f / l_run[j]);
            O[obase + (size_t)(j + half * 8) * DHEAD + db * 16 + l16] = o;
        }
}

extern "C" void kernel_launch(void* const* d_in, const int* in_sizes, int n_in,
                              void* d_out, int out_size, void* d_ws, size_t ws_size,
                              hipStream_t stream) {
    const float* Q = (const float*)d_in[0];
    const float* K = (const float*)d_in[1];
    const float* V = (const float*)d_in[2];
    float*       O = (float*)d_out;
    (void)in_sizes; (void)n_in; (void)out_size; (void)ws_size;

    const size_t elems = (size_t)NHEAD * S_LEN * DHEAD;     // 4,194,304 per tensor
    _Float16* Kh  = (_Float16*)d_ws;                        // 8.4 MB
    _Float16* VhT = Kh + elems;                             // 8.4 MB (transposed per head)

    cvt_f16_kernel<<<dim3((unsigned)(elems / 8 / 256)), dim3(256), 0, stream>>>(K, Kh, (int)(elems / 8));
    vtrans_kernel<<<dim3(S_LEN / 64, NHEAD), dim3(256), 0, stream>>>(V, VhT);
    fa_wmma_kernel<<<dim3(S_LEN / BR, NHEAD), dim3(128), 0, stream>>>(Q, Kh, VhT, O);
}
